// PixelGNN_80161269613205
// MI455X (gfx1250) — matrix operand order repaired and checked by reading, not compiled, not run on hardware
//
#include <hip/hip_runtime.h>
#include <hip/hip_bf16.h>

typedef __attribute__((ext_vector_type(16))) _Float16 v16h;
typedef __attribute__((ext_vector_type(8)))  _Float16 v8h;
typedef __attribute__((ext_vector_type(8)))  float    v8f;

union V16H { v16h v; v8h h[2]; };
union V8F  { v8f v; float f[8]; };

__device__ __forceinline__ float wsum(float v) {
#pragma unroll
  for (int off = 16; off > 0; off >>= 1) v += __shfl_xor(v, off, 32);
  return v;
}

__device__ __forceinline__ void atomicMaxF(float* addr, float val) {
  int old = __float_as_int(*addr);
  while (__int_as_float(old) < val) {
    int prev = atomicCAS((int*)addr, old, __float_as_int(val));
    if (prev == old) break;
    old = prev;
  }
}

// ---------------- utility kernels ----------------

__global__ void k_fill(float* p, float v, int n) {
  int i = blockIdx.x * blockDim.x + threadIdx.x;
  if (i < n) p[i] = v;
}

__global__ void k_f32_to_f16(const float* __restrict__ s, _Float16* __restrict__ d, int n) {
  int i = blockIdx.x * blockDim.x + threadIdx.x;
  if (i < n) d[i] = (_Float16)s[i];
}

// ---------------- encoder: one wave per node, lane = channel ----------------

__global__ __launch_bounds__(256) void k_encode(
    const float* __restrict__ samples, const float* __restrict__ w1,
    const float* __restrict__ b1, const float* __restrict__ w2,
    const float* __restrict__ b2, const float* __restrict__ lnw,
    const float* __restrict__ lnb, _Float16* __restrict__ x16, int Nn, int S) {
  int lane = threadIdx.x & 31;
  int node = (blockIdx.x * blockDim.x + threadIdx.x) >> 5;
  if (node >= Nn) return;  // wave-uniform
  const float* sp = samples + (size_t)node * S * 2;
  float w10 = w1[lane * 2 + 0], w11 = w1[lane * 2 + 1], bb1 = b1[lane];
  float w2r[32];
#pragma unroll
  for (int k = 0; k < 32; k++) w2r[k] = w2[lane * 32 + k];
  float bb2 = b2[lane], lw = lnw[lane], lb = lnb[lane];
  float acc = 0.f, cnt = 0.f;
  for (int s = 0; s < S; s++) {
    float s0 = sp[s * 2 + 0], s1 = sp[s * 2 + 1];
    float h1 = fmaxf(w10 * (s0 * 1e-4f) + w11 * (s1 * 5e-5f) + bb1, 0.f);
    float h2 = bb2;
#pragma unroll
    for (int k = 0; k < 32; k++) h2 += w2r[k] * __shfl(h1, k, 32);
    h2 = fmaxf(h2, 0.f);
    float mu = wsum(h2) * (1.f / 32.f);
    float d = h2 - mu;
    float var = wsum(d * d) * (1.f / 32.f);
    float xn = d * rsqrtf(var + 1e-5f) * lw + lb;
    float m = (fabsf(s0) + fabsf(s1)) > 0.f ? 1.f : 0.f;
    acc += xn * m;
    cnt += m;
  }
  x16[(size_t)node * 32 + lane] = (_Float16)(acc / fmaxf(cnt, 1e-6f));
}

// ---------------- edge-attr mean over dst ----------------

__global__ void k_edge_stats(const int* __restrict__ dst, const float* __restrict__ ea,
                             float* __restrict__ cnt, float* __restrict__ eas, int E) {
  int e = blockIdx.x * blockDim.x + threadIdx.x;
  if (e < E) {
    int d = dst[e];
    atomicAdd(&cnt[d], 1.f);
    atomicAdd(&eas[d], ea[e]);
  }
}

__global__ void k_eam(const float* __restrict__ eas, const float* __restrict__ cnt,
                      float* __restrict__ eam, int Nn) {
  int i = blockIdx.x * blockDim.x + threadIdx.x;
  if (i < Nn) eam[i] = eas[i] / fmaxf(cnt[i], 1.f);
}

// ---------------- WMMA GEMM: Y[N x Mout] = X[N x K](f16) @ W[Mout x K]^T(f16) + b ----------------
// One wave computes one 16x16 output tile with v_wmma_f32_16x16x32_f16.
// A lane layout (16-bit A, 16x32): lane group g = lane>>4; halves 0..15 -> K = 16g..16g+15
// (two contiguous 16B loads).
// B lane layout (16-bit B, 32x16): lane holds column n = lane&15, K = 16g..16g+15 (one 32B load).

__global__ __launch_bounds__(256) void k_gemm_wmma(
    const _Float16* __restrict__ X, const _Float16* __restrict__ Wt,
    const float* __restrict__ bias, float* __restrict__ Y,
    int K, int Mout, int totalTiles) {
  int lane = threadIdx.x & 31;
  int wave = threadIdx.x >> 5;
  int tile = blockIdx.x * 8 + wave;
  if (tile >= totalTiles) return;  // wave-uniform; EXEC all-1s inside
  int tilesN = Mout >> 4;
  int tm = tile / tilesN;
  int tn = tile - tm * tilesN;
  int lh = lane >> 4;
  int ll = lane & 15;
  int row = tm * 16 + ll;
  int col = tn * 16 + ll;

  V8F acc;
  float bn = bias[col];
#pragma unroll
  for (int r = 0; r < 8; r++) acc.f[r] = bn;
  v8f c = acc.v;

  int nk = K >> 5;
  for (int kc = 0; kc < nk; kc++) {
    V16H a, b;
    const _Float16* ap = X + (size_t)row * K + kc * 32 + 16 * lh;
    a.h[0] = *(const v8h*)(ap);
    a.h[1] = *(const v8h*)(ap + 8);
    const _Float16* bp = Wt + (size_t)col * K + kc * 32 + 16 * lh;
    b.v = *(const v16h*)(bp);
    c = __builtin_amdgcn_wmma_f32_16x16x32_f16(false, a.v, false, b.v,
                                               (short)0, c, false, false);
  }
  acc.v = c;
  float* yp = Y + (size_t)(tm * 16 + 8 * lh) * (size_t)Mout + col;
#pragma unroll
  for (int r = 0; r < 8; r++) yp[(size_t)r * Mout] = acc.f[r];
}

// ---------------- edge pass 1: logits + segment max (one wave per edge) ----------------

__global__ __launch_bounds__(256) void k_edge_logit_max(
    const int* __restrict__ src, const int* __restrict__ dst,
    const float* __restrict__ ea, const float* __restrict__ eam,
    const float* __restrict__ XL, const float* __restrict__ XR,
    const float* __restrict__ we, const float* __restrict__ att,
    float* __restrict__ elog, float* __restrict__ maxv, int E, int Nn) {
  int lane = threadIdx.x & 31;
  int wid = (blockIdx.x * blockDim.x + threadIdx.x) >> 5;
  int E2 = E + Nn;
  if (wid >= E2) return;
  int s_, d_;
  float eav;
  if (wid < E) { s_ = src[wid]; d_ = dst[wid]; eav = ea[wid]; }
  else         { s_ = d_ = wid - E; eav = eam[wid - E]; }
  const float* xlp = XL + (size_t)s_ * 256;
  const float* xrp = XR + (size_t)d_ * 256;
  __builtin_prefetch(xlp, 0, 1);   // global_prefetch_b8
  __builtin_prefetch(xrp, 0, 1);
#pragma unroll
  for (int h = 0; h < 4; h++) {
    float p = 0.f;
#pragma unroll
    for (int cc = 0; cc < 2; cc++) {
      int j = h * 64 + lane + cc * 32;
      float z = xlp[j] + xrp[j] + eav * we[j];
      z = z > 0.f ? z : 0.2f * z;   // LeakyReLU(0.2)
      p += att[j] * z;
    }
    p = wsum(p);
    if (lane == 0) {
      elog[(size_t)wid * 4 + h] = p;
      atomicMaxF(&maxv[(size_t)d_ * 4 + h], p);
    }
  }
}

// ---------------- edge pass 2: exp + denom ----------------

__global__ void k_edge_exp(const int* __restrict__ dst, float* __restrict__ elog,
                           const float* __restrict__ maxv, float* __restrict__ denom,
                           int E, int Nn) {
  int idx = blockIdx.x * blockDim.x + threadIdx.x;
  int E2 = E + Nn;
  if (idx >= E2 * 4) return;
  int e = idx >> 2, h = idx & 3;
  int d_ = (e < E) ? dst[e] : (e - E);
  float a = __expf(elog[idx] - maxv[(size_t)d_ * 4 + h]);
  elog[idx] = a;
  atomicAdd(&denom[(size_t)d_ * 4 + h], a);
}

// ---------------- edge pass 3: weighted scatter-add of xl[src] ----------------

__global__ __launch_bounds__(256) void k_edge_acc(
    const int* __restrict__ src, const int* __restrict__ dst,
    const float* __restrict__ elog, const float* __restrict__ denom,
    const float* __restrict__ XL, float* __restrict__ OUT, int E, int Nn) {
  int lane = threadIdx.x & 31;
  int wid = (blockIdx.x * blockDim.x + threadIdx.x) >> 5;
  int E2 = E + Nn;
  if (wid >= E2) return;
  int s_, d_;
  if (wid < E) { s_ = src[wid]; d_ = dst[wid]; }
  else         { s_ = d_ = wid - E; }
  const float* xlp = XL + (size_t)s_ * 256;
  float* op = OUT + (size_t)d_ * 256;
#pragma unroll
  for (int h = 0; h < 4; h++) {
    float w = elog[(size_t)wid * 4 + h] / (denom[(size_t)d_ * 4 + h] + 1e-16f);
#pragma unroll
    for (int cc = 0; cc < 2; cc++) {
      int j = h * 64 + lane + cc * 32;
      atomicAdd(&op[j], xlp[j] * w);
    }
  }
}

// ---------------- node finalize: head-mean + bias + relu (+ optional output head) ----------------

__global__ __launch_bounds__(256) void k_finalize(
    const float* __restrict__ OUT, const float* __restrict__ bias,
    _Float16* __restrict__ x16, const float* __restrict__ hw,
    const float* __restrict__ hb, float* __restrict__ outp, int Nn, int doHead) {
  int lane = threadIdx.x & 31;
  int i = (blockIdx.x * blockDim.x + threadIdx.x) >> 5;
  if (i >= Nn) return;
  const float* op = OUT + (size_t)i * 256;
  float v0 = 0.f, v1 = 0.f;
#pragma unroll
  for (int h = 0; h < 4; h++) {
    v0 += op[h * 64 + lane];
    v1 += op[h * 64 + lane + 32];
  }
  v0 = fmaxf(0.25f * v0 + bias[lane], 0.f);
  v1 = fmaxf(0.25f * v1 + bias[lane + 32], 0.f);
  x16[(size_t)i * 64 + lane] = (_Float16)v0;
  x16[(size_t)i * 64 + lane + 32] = (_Float16)v1;
  if (doHead) {
    float p0 = hw[lane] * v0 + hw[lane + 32] * v1;
    float p1 = hw[64 + lane] * v0 + hw[96 + lane] * v1;
    p0 = wsum(p0);
    p1 = wsum(p1);
    if (lane == 0) {
      outp[i] = p0 + hb[0];
      float lv = p1 + hb[1];
      outp[Nn + i] = fminf(fmaxf(lv, -5.f), 10.f);
    }
  }
}

// ---------------- launcher ----------------

extern "C" void kernel_launch(void* const* d_in, const int* in_sizes, int n_in,
                              void* d_out, int out_size, void* d_ws, size_t ws_size,
                              hipStream_t stream) {
  const float* samples   = (const float*)d_in[0];
  const int*   edge_idx  = (const int*)d_in[1];
  const float* edge_attr = (const float*)d_in[2];
  const float* e_fc1_w = (const float*)d_in[3];
  const float* e_fc1_b = (const float*)d_in[4];
  const float* e_fc2_w = (const float*)d_in[5];
  const float* e_fc2_b = (const float*)d_in[6];
  const float* e_ln_w  = (const float*)d_in[7];
  const float* e_ln_b  = (const float*)d_in[8];
  const float* gw[2][7] = {
    {(const float*)d_in[9],  (const float*)d_in[10], (const float*)d_in[11],
     (const float*)d_in[12], (const float*)d_in[13], (const float*)d_in[14],
     (const float*)d_in[15]},
    {(const float*)d_in[16], (const float*)d_in[17], (const float*)d_in[18],
     (const float*)d_in[19], (const float*)d_in[20], (const float*)d_in[21],
     (const float*)d_in[22]}};
  const float* head_w = (const float*)d_in[23];
  const float* head_b = (const float*)d_in[24];

  const int S = 50;
  const int N = in_sizes[0] / (S * 2);
  const int E = in_sizes[1] / 2;
  const int E2 = E + N;
  const int* src = edge_idx;
  const int* dst = edge_idx + E;

  // workspace carve-up (256B aligned regions)
  char* wsb = (char*)d_ws;
  size_t off = 0;
  auto carve = [&](size_t bytes) -> void* {
    void* p = wsb + off;
    off = (off + bytes + 255) & ~(size_t)255;
    return p;
  };
  _Float16* wl16 = (_Float16*)carve((size_t)256 * 64 * 2);
  _Float16* wr16 = (_Float16*)carve((size_t)256 * 64 * 2);
  _Float16* x16  = (_Float16*)carve((size_t)N * 64 * 2);
  float* XL   = (float*)carve((size_t)N * 256 * 4);
  float* XR   = (float*)carve((size_t)N * 256 * 4);  // doubles as OUT accumulator
  float* ELOG = (float*)carve((size_t)E2 * 4 * 4);
  float* MAXV = (float*)carve((size_t)N * 4 * 4);
  float* DEN  = (float*)carve((size_t)N * 4 * 4);
  float* EAM  = (float*)carve((size_t)N * 4);
  float* CNT  = (float*)carve((size_t)N * 4);
  float* EAS  = (float*)carve((size_t)N * 4);

  const int TB = 256;
  int nodeWaveBlocks = (N + 7) / 8;
  int edgeWaveBlocks = (E2 + 7) / 8;

  // encoder
  k_encode<<<nodeWaveBlocks, TB, 0, stream>>>(samples, e_fc1_w, e_fc1_b, e_fc2_w,
                                              e_fc2_b, e_ln_w, e_ln_b, x16, N, S);
  // edge-attr mean
  k_fill<<<(N + TB - 1) / TB, TB, 0, stream>>>(CNT, 0.f, N);
  k_fill<<<(N + TB - 1) / TB, TB, 0, stream>>>(EAS, 0.f, N);
  k_edge_stats<<<(E + TB - 1) / TB, TB, 0, stream>>>(dst, edge_attr, CNT, EAS, E);
  k_eam<<<(N + TB - 1) / TB, TB, 0, stream>>>(EAS, CNT, EAM, N);

  for (int layer = 0; layer < 2; layer++) {
    const float* wl = gw[layer][0];
    const float* bl = gw[layer][1];
    const float* wr = gw[layer][2];
    const float* br = gw[layer][3];
    const float* we = gw[layer][4];
    const float* att = gw[layer][5];
    const float* gbias = gw[layer][6];
    int K = layer ? 64 : 32;
    int wn = 256 * K;

    // weight f32 -> f16
    k_f32_to_f16<<<(wn + TB - 1) / TB, TB, 0, stream>>>(wl, wl16, wn);
    k_f32_to_f16<<<(wn + TB - 1) / TB, TB, 0, stream>>>(wr, wr16, wn);

    // dense linears via WMMA
    int tiles = ((N + 15) / 16) * (256 / 16);
    int gemmBlocks = (tiles + 7) / 8;
    k_gemm_wmma<<<gemmBlocks, TB, 0, stream>>>(x16, wl16, bl, XL, K, 256, tiles);
    k_gemm_wmma<<<gemmBlocks, TB, 0, stream>>>(x16, wr16, br, XR, K, 256, tiles);

    // attention passes
    k_fill<<<(N * 4 + TB - 1) / TB, TB, 0, stream>>>(MAXV, -3.402823466e38f, N * 4);
    k_edge_logit_max<<<edgeWaveBlocks, TB, 0, stream>>>(src, dst, edge_attr, EAM, XL,
                                                        XR, we, att, ELOG, MAXV, E, N);
    k_fill<<<(N * 4 + TB - 1) / TB, TB, 0, stream>>>(DEN, 0.f, N * 4);
    k_edge_exp<<<(E2 * 4 + TB - 1) / TB, TB, 0, stream>>>(dst, ELOG, MAXV, DEN, E, N);

    // XR is dead (logits stored) -> reuse as the output accumulator
    k_fill<<<(N * 256 + TB - 1) / TB, TB, 0, stream>>>(XR, 0.f, N * 256);
    k_edge_acc<<<edgeWaveBlocks, TB, 0, stream>>>(src, dst, ELOG, DEN, XL, XR, E, N);

    // head-mean + bias + relu; last layer also applies the output head
    k_finalize<<<nodeWaveBlocks, TB, 0, stream>>>(XR, gbias, x16, head_w, head_b,
                                                  (float*)d_out, N, layer == 1);
  }
}
